// MeshedDecoder_40742059770532
// MI455X (gfx1250) — matrix operand order repaired and checked by reading, not compiled, not run on hardware
//
#include <hip/hip_runtime.h>

// ---------------------------------------------------------------------------
// MeshedDecoder forward for MI455X (gfx1250), all GEMMs via v_wmma_f32_16x16x32_bf16
// ---------------------------------------------------------------------------

typedef __attribute__((ext_vector_type(16))) __bf16 v16bf;
typedef __attribute__((ext_vector_type(8)))  float  v8f;

namespace {
constexpr int cB  = 64;
constexpr int cS  = 32;
constexpr int cNE = 3;
constexpr int cNF = 196;
constexpr int cD  = 512;
constexpr int cH  = 8;
constexpr int cDK = 64;
constexpr int cFF = 2048;
constexpr int cL  = 3;
constexpr int cV  = 15000;
constexpr float cNEG = -1000000000.0f;
constexpr int cBT = cB * cS;          // 2048 token rows
constexpr int cFR = cB * cNE * cNF;   // 37632 feature rows
constexpr int cER = cB * cNE * cS;    // 6144 enc rows
}

__device__ __forceinline__ __bf16 f2bf(float f) {
  union { float f; unsigned u; } c; c.f = f;
  unsigned r = c.u + 0x7FFFu + ((c.u >> 16) & 1u);
  union { unsigned short s; __bf16 b; } o;
  o.s = (unsigned short)(r >> 16);
  return o.b;
}

// ---------------------------------------------------------------------------
// WMMA GEMM: C[M,N] = act(A[M,K] (f32) x Bt[N,K] (bf16, pre-transposed) + bias)
// block = 256 threads = 8 waves; tile 128x128x32; wave tile 64x32 (4x2 wmma).
// Double-buffered LDS (one barrier per K-step), global loads for tile k+1
// issued before the WMMA block of tile k, GL2 prefetch two tiles ahead.
// Requires K % 32 == 0 (true for all call sites: 512/1024/2048).
// ---------------------------------------------------------------------------
#define GBM 128
#define GBN 128
#define GBK 32
#define GPITCH 48   // bf16 elems per LDS row (96B: 16B aligned, non-pow2 bank stride)

union Frag16 { uint4 u[2]; v16bf v; };

__global__ __launch_bounds__(256) void md_gemm_kernel(
    const float*  __restrict__ A,
    const __bf16* __restrict__ Bt,
    const float*  __restrict__ bias,
    float*        __restrict__ C,
    int M, int N, int K,
    long sA, long sB, long sBias, long sC,
    int act) // 0=none 1=relu 2=sigmoid
{
  __shared__ __bf16 As[2][GBM][GPITCH];
  __shared__ __bf16 Bs[2][GBN][GPITCH];

  const int tid  = threadIdx.x;
  const int lane = tid & 31;
  const int wave = tid >> 5;
  const int wm   = wave >> 2;   // 0..1 (64 rows each)
  const int wn   = wave & 3;    // 0..3 (32 cols each)
  const int m0   = blockIdx.y * GBM;
  const int n0   = blockIdx.x * GBN;
  const long z   = blockIdx.z;

  A  += z * sA;
  Bt += z * sB;
  C  += z * sC;
  const float* biasp = bias ? (bias + z * sBias) : nullptr;

  v8f acc[4][2] = {};

  const int  lrow  = tid >> 1;        // 0..127
  const int  lhalf = (tid & 1) * 16;  // 0 or 16 (K elements)
  const int  am    = m0 + lrow;
  const bool aok   = am < M;
  const int  bn    = n0 + lrow;
  const bool bok   = bn < N;
  const float*  aBase = A  + (long)(aok ? am : 0) * K + lhalf;
  const __bf16* bBase = Bt + (long)(bok ? bn : 0) * K + lhalf;

  // ---- staged global loads (tile k) into registers ----
  auto loadA_g = [&](int k0, float4 af[4]) {
    if (aok) {
      const float* ap = aBase + k0;
#pragma unroll
      for (int i = 0; i < 4; i++) af[i] = *(const float4*)(ap + 4 * i);
    } else {
#pragma unroll
      for (int i = 0; i < 4; i++) af[i] = make_float4(0.f, 0.f, 0.f, 0.f);
    }
  };
  auto loadB_g = [&](int k0, uint4& b0, uint4& b1) {
    if (bok) {
      const __bf16* bp = bBase + k0;
      b0 = *(const uint4*)bp;
      b1 = *(const uint4*)(bp + 8);
    } else {
      b0 = uint4{0, 0, 0, 0};
      b1 = uint4{0, 0, 0, 0};
    }
  };
  // ---- register -> LDS stores ----
  auto storeA = [&](int buf, const float4 af[4]) {
    __align__(16) __bf16 tmp[16];
#pragma unroll
    for (int i = 0; i < 4; i++) {
      tmp[4 * i + 0] = f2bf(af[i].x); tmp[4 * i + 1] = f2bf(af[i].y);
      tmp[4 * i + 2] = f2bf(af[i].z); tmp[4 * i + 3] = f2bf(af[i].w);
    }
    *(uint4*)&As[buf][lrow][lhalf]     = *(const uint4*)&tmp[0];
    *(uint4*)&As[buf][lrow][lhalf + 8] = *(const uint4*)&tmp[8];
  };
  auto storeB = [&](int buf, const uint4& b0, const uint4& b1) {
    *(uint4*)&Bs[buf][lrow][lhalf]     = b0;
    *(uint4*)&Bs[buf][lrow][lhalf + 8] = b1;
  };
  // ---- WMMA on one LDS buffer ----
  auto compute = [&](int buf) {
    Frag16 fa[4], fb[2];
    const int l15  = lane & 15;
    const int ksel = (lane >> 4) * 8;   // A: lanes>=16 hold K 8..15 / 24..31
#pragma unroll
    for (int i = 0; i < 4; i++) {
      const int r = wm * 64 + i * 16 + l15;
      fa[i].u[0] = *(const uint4*)&As[buf][r][ksel];
      fa[i].u[1] = *(const uint4*)&As[buf][r][16 + ksel];
    }
    const int ksel2 = (lane >> 4) * 8;  // B: lanes>=16 hold K 8..15 / 24..31
#pragma unroll
    for (int j = 0; j < 2; j++) {
      const int n = wn * 32 + j * 16 + l15;
      fb[j].u[0] = *(const uint4*)&Bs[buf][n][ksel2];
      fb[j].u[1] = *(const uint4*)&Bs[buf][n][16 + ksel2];
    }
#pragma unroll
    for (int i = 0; i < 4; i++)
#pragma unroll
      for (int j = 0; j < 2; j++)
        acc[i][j] = __builtin_amdgcn_wmma_f32_16x16x32_bf16(
            false, fa[i].v, false, fb[j].v, (short)0, acc[i][j], false, false);
  };

  const int KT = K / GBK;

  // prologue: stage tile 0 into buffer 0
  {
    float4 af[4]; uint4 b0, b1;
    loadA_g(0, af);
    loadB_g(0, b0, b1);
    storeA(0, af);
    storeB(0, b0, b1);
  }
  __syncthreads();

  for (int kt = 0; kt < KT; kt++) {
    const int cur = kt & 1;
    const int nxt = cur ^ 1;
    const bool haveNext = (kt + 1) < KT;

    float4 af[4]; uint4 b0, b1;
    if (haveNext) {
      const int k0 = (kt + 1) * GBK;
      // warm GL2 two tiles ahead (global_prefetch_b8)
      if (kt + 2 < KT) {
        __builtin_prefetch(aBase + (kt + 2) * GBK, 0, 1);
        __builtin_prefetch(bBase + (kt + 2) * GBK, 0, 1);
      }
      loadA_g(k0, af);          // global loads in flight during WMMA below
      loadB_g(k0, b0, b1);
    }

    compute(cur);               // 8x v_wmma_f32_16x16x32_bf16

    if (haveNext) {
      storeA(nxt, af);
      storeB(nxt, b0, b1);
    }
    __syncthreads();
  }

  // ---- epilogue: C layout = VGPR j -> M=j (lanes<16) / j+8 (lanes>=16) ----
#pragma unroll
  for (int i = 0; i < 4; i++) {
#pragma unroll
    for (int j = 0; j < 2; j++) {
#pragma unroll
      for (int v = 0; v < 8; v++) {
        const int row = m0 + wm * 64 + i * 16 + ((lane < 16) ? v : v + 8);
        const int col = n0 + wn * 32 + j * 16 + (lane & 15);
        if (row < M && col < N) {
          float val = acc[i][j][v];
          if (biasp) val += biasp[col];
          if (act == 1)      val = fmaxf(val, 0.0f);
          else if (act == 2) val = 1.0f / (1.0f + __expf(-val));
          C[(long)row * N + col] = val;
        }
      }
    }
  }
}

// ---------------------------------------------------------------------------
// Weight convert+transpose: W[mat][K][N] f32 -> Bt[mat][N][K] bf16
// ---------------------------------------------------------------------------
__global__ void md_convert_wt(const float* __restrict__ W, __bf16* __restrict__ Bt,
                              int K, int N, long total) {
  long idx = (long)blockIdx.x * blockDim.x + threadIdx.x;
  if (idx >= total) return;
  const long nk  = (long)N * K;
  const long mat = idx / nk;
  const long mk  = idx % nk;
  const int n = (int)(mk / K);
  const int k = (int)(mk % K);
  Bt[idx] = f2bf(W[mat * nk + (long)k * N + n]);
}

// ---------------------------------------------------------------------------
// Elementwise / attention helper kernels
// ---------------------------------------------------------------------------
__global__ void md_sinusoid(float* __restrict__ t, int n, int d, int zero0) {
  int idx = blockIdx.x * blockDim.x + threadIdx.x;
  if (idx >= n * d) return;
  const int p = idx / d, i = idx % d;
  const float ang = (float)p * __powf(10000.0f, -(float)(2 * (i / 2)) / (float)d);
  float v = (i & 1) ? __cosf(ang) : __sinf(ang);
  if (zero0 && p == 0) v = 0.0f;
  t[idx] = v;
}

__global__ void md_embed(const int* __restrict__ tok, const float* __restrict__ emb,
                         float* __restrict__ out) {
  long idx = (long)blockIdx.x * blockDim.x + threadIdx.x;
  if (idx >= (long)cBT * cD) return;
  const long r = idx / cD;
  const int d  = (int)(idx % cD);
  out[idx] = emb[(long)tok[r] * cD + d];
}

__global__ void md_addpos(const float* __restrict__ out, const float* __restrict__ tokpos,
                          const int* __restrict__ tok, float* __restrict__ x) {
  long idx = (long)blockIdx.x * blockDim.x + threadIdx.x;
  if (idx >= (long)cBT * cD) return;
  const long r = idx / cD;
  const int d  = (int)(idx % cD);
  const int q  = (int)(r % cS);
  const int row = (tok[r] == 0) ? 0 : (q + 1);
  x[idx] = out[idx] + tokpos[row * cD + d];
}

__global__ void md_feats(const float* __restrict__ features, const float* __restrict__ fpos,
                         float* __restrict__ feats) {
  long idx = (long)blockIdx.x * blockDim.x + threadIdx.x;
  if (idx >= (long)cFR * cD) return;
  const int d = (int)(idx % cD);
  const int f = (int)((idx / cD) % cNF);
  feats[idx] = features[idx] + fpos[f * cD + d];
}

__global__ void md_self_scores(const float* __restrict__ q, const float* __restrict__ k,
                               const int* __restrict__ tok, float* __restrict__ s) {
  long idx = (long)blockIdx.x * blockDim.x + threadIdx.x;
  if (idx >= (long)cB * cH * cS * cS) return;
  const int kk = (int)(idx % cS);
  const int qq = (int)((idx / cS) % cS);
  const int h  = (int)((idx / ((long)cS * cS)) % cH);
  const int b  = (int)(idx / ((long)cS * cS * cH));
  float val;
  if (kk > qq || tok[b * cS + kk] == 0) {
    val = cNEG;
  } else {
    const float* qp = q + ((long)(b * cS + qq)) * cD + h * cDK;
    const float* kp = k + ((long)(b * cS + kk)) * cD + h * cDK;
    float dot = 0.f;
#pragma unroll 8
    for (int d = 0; d < cDK; d++) dot += qp[d] * kp[d];
    val = dot * 0.125f;
  }
  s[idx] = val;
}

__global__ void md_cross_scores(const float* __restrict__ qe, const float* __restrict__ ke,
                                const unsigned char* __restrict__ fmask, float* __restrict__ s) {
  long idx = (long)blockIdx.x * blockDim.x + threadIdx.x;
  if (idx >= (long)cB * cNE * cH * cS * cNF) return;
  const int kk = (int)(idx % cNF);
  long t = idx / cNF;
  const int qq = (int)(t % cS); t /= cS;
  const int h  = (int)(t % cH); t /= cH;
  const int e  = (int)(t % cNE);
  const int b  = (int)(t / cNE);
  float val;
  if (fmask[b * cNF + kk]) {
    val = cNEG;
  } else {
    const float* qp = qe + ((long)(b * cS + qq)) * cD + h * cDK;
    const float* kp = ke + ((long)((b * cNE + e) * cNF + kk)) * cD + h * cDK;
    float dot = 0.f;
#pragma unroll 8
    for (int d = 0; d < cDK; d++) dot += qp[d] * kp[d];
    val = dot * 0.125f;
  }
  s[idx] = val;
}

// one wave32 per row
__global__ void md_softmax_rows(float* __restrict__ s, int rows, int len) {
  const int wave = blockIdx.x * (blockDim.x >> 5) + (threadIdx.x >> 5);
  const int lane = threadIdx.x & 31;
  if (wave >= rows) return;
  float* row = s + (long)wave * len;
  float m = -3.4e38f;
  for (int i = lane; i < len; i += 32) m = fmaxf(m, row[i]);
  for (int o = 16; o; o >>= 1) m = fmaxf(m, __shfl_xor(m, o, 32));
  float sum = 0.f;
  for (int i = lane; i < len; i += 32) { float e = __expf(row[i] - m); row[i] = e; sum += e; }
  for (int o = 16; o; o >>= 1) sum += __shfl_xor(sum, o, 32);
  const float inv = 1.0f / sum;
  for (int i = lane; i < len; i += 32) row[i] *= inv;
}

__global__ void md_self_av(const float* __restrict__ a, const float* __restrict__ v,
                           float* __restrict__ o) {
  long idx = (long)blockIdx.x * blockDim.x + threadIdx.x;
  if (idx >= (long)cBT * cD) return;
  const int c = (int)(idx % cD);
  const long r = idx / cD;            // b*S+q
  const int q = (int)(r % cS);
  const int b = (int)(r / cS);
  const int h = c / cDK;
  const float* ap = a + (((long)b * cH + h) * cS + q) * cS;
  const float* vp = v + (long)b * cS * cD + c;
  float sum = 0.f;
#pragma unroll 8
  for (int kk = 0; kk < cS; kk++) sum += ap[kk] * vp[(long)kk * cD];
  o[idx] = sum;
}

__global__ void md_cross_av(const float* __restrict__ a, const float* __restrict__ ve,
                            float* __restrict__ o) {
  long idx = (long)blockIdx.x * blockDim.x + threadIdx.x;
  if (idx >= (long)cER * cD) return;
  const int c = (int)(idx % cD);
  const long r = idx / cD;            // (b*NE+e)*S+q
  const int q = (int)(r % cS);
  const long be = r / cS;             // b*NE+e
  const int h = c / cDK;
  const float* ap = a + ((be * cH + h) * cS + q) * (long)cNF;
  const float* vp = ve + be * (long)cNF * cD + c;
  float sum = 0.f;
#pragma unroll 4
  for (int kk = 0; kk < cNF; kk++) sum += ap[kk] * vp[(long)kk * cD];
  o[idx] = sum;
}

// out = padzero? 0 : LN(A + B) * g + beta   (block per row, D=512)
__global__ __launch_bounds__(256) void md_ln(
    const float* __restrict__ Abuf, const float* __restrict__ Bbuf,
    const float* __restrict__ g, const float* __restrict__ beta,
    float* __restrict__ out, int D, int bcastA, int doPad,
    const int* __restrict__ tok) {
  const int r = blockIdx.x;
  int arow = r;
  if (bcastA) {
    const int b = r / (cNE * cS);
    const int q = r % cS;
    arow = b * cS + q;
  }
  const float* ap = Abuf + (long)arow * D;
  const float* bp = Bbuf + (long)r * D;
  float* op = out + (long)r * D;
  const bool pad = doPad && (tok[arow] == 0);

  __shared__ float red[256];
  float s = 0.f;
  for (int i = threadIdx.x; i < D; i += 256) s += ap[i] + bp[i];
  red[threadIdx.x] = s; __syncthreads();
  for (int st = 128; st > 0; st >>= 1) {
    if (threadIdx.x < st) red[threadIdx.x] += red[threadIdx.x + st];
    __syncthreads();
  }
  const float mean = red[0] / (float)D;
  __syncthreads();
  float v = 0.f;
  for (int i = threadIdx.x; i < D; i += 256) {
    const float t = ap[i] + bp[i] - mean;
    v += t * t;
  }
  red[threadIdx.x] = v; __syncthreads();
  for (int st = 128; st > 0; st >>= 1) {
    if (threadIdx.x < st) red[threadIdx.x] += red[threadIdx.x + st];
    __syncthreads();
  }
  const float inv = rsqrtf(red[0] / (float)D + 1e-5f);
  for (int i = threadIdx.x; i < D; i += 256) {
    const float t = (ap[i] + bp[i] - mean) * inv * g[i] + beta[i];
    op[i] = pad ? 0.0f : t;
  }
}

// cat[e][b*S+q][0:D]=sa, [D:2D]=enc
__global__ void md_cat(const float* __restrict__ sa, const float* __restrict__ enc,
                       float* __restrict__ cat) {
  long idx = (long)blockIdx.x * blockDim.x + threadIdx.x;
  if (idx >= (long)cNE * cBT * 2 * cD) return;
  const int f = (int)(idx % (2 * cD));
  const long t = idx / (2 * cD);
  const int r = (int)(t % cBT);   // b*S+q
  const int e = (int)(t / cBT);
  float v;
  if (f < cD) {
    v = sa[(long)r * cD + f];
  } else {
    const int b = r / cS, q = r % cS;
    v = enc[((long)(b * cNE + e) * cS + q) * cD + (f - cD)];
  }
  cat[idx] = v;
}

__global__ void md_combine(const float* __restrict__ enc, const float* __restrict__ alpha,
                           float* __restrict__ attn) {
  long idx = (long)blockIdx.x * blockDim.x + threadIdx.x;
  if (idx >= (long)cBT * cD) return;
  const int d = (int)(idx % cD);
  const int r = (int)(idx / cD);  // b*S+q
  const int b = r / cS, q = r % cS;
  float s = 0.f;
#pragma unroll
  for (int e = 0; e < cNE; e++) {
    const float ev = enc[((long)(b * cNE + e) * cS + q) * cD + d];
    const float av = alpha[((long)e * cBT + r) * cD + d];
    s += ev * av;
  }
  attn[idx] = s * rsqrtf((float)cNE);
}

__global__ __launch_bounds__(256) void md_logsoftmax(float* __restrict__ x, int len) {
  float* row = x + (long)blockIdx.x * len;
  __shared__ float red[256];
  float m = -3.4e38f;
  for (int i = threadIdx.x; i < len; i += 256) m = fmaxf(m, row[i]);
  red[threadIdx.x] = m; __syncthreads();
  for (int st = 128; st > 0; st >>= 1) {
    if (threadIdx.x < st) red[threadIdx.x] = fmaxf(red[threadIdx.x], red[threadIdx.x + st]);
    __syncthreads();
  }
  m = red[0]; __syncthreads();
  float sum = 0.f;
  for (int i = threadIdx.x; i < len; i += 256) sum += __expf(row[i] - m);
  red[threadIdx.x] = sum; __syncthreads();
  for (int st = 128; st > 0; st >>= 1) {
    if (threadIdx.x < st) red[threadIdx.x] += red[threadIdx.x + st];
    __syncthreads();
  }
  const float lse = m + __logf(red[0]);
  __syncthreads();
  for (int i = threadIdx.x; i < len; i += 256) row[i] -= lse;
}

// ---------------------------------------------------------------------------
// Host orchestration
// ---------------------------------------------------------------------------
extern "C" void kernel_launch(void* const* d_in, const int* in_sizes, int n_in,
                              void* d_out, int out_size, void* d_ws, size_t ws_size,
                              hipStream_t stream) {
  (void)in_sizes; (void)n_in; (void)out_size; (void)ws_size;

  const int*           tok      = (const int*)d_in[0];
  const float*         features = (const float*)d_in[1];
  const unsigned char* fmask    = (const unsigned char*)d_in[2];
  const float*         emb      = (const float*)d_in[3];
  const float* Wq_s   = (const float*)d_in[4];
  const float* bq_s   = (const float*)d_in[5];
  const float* Wk_s   = (const float*)d_in[6];
  const float* bk_s   = (const float*)d_in[7];
  const float* Wv_s   = (const float*)d_in[8];
  const float* bv_s   = (const float*)d_in[9];
  const float* Wo_s   = (const float*)d_in[10];
  const float* bo_s   = (const float*)d_in[11];
  const float* g_s    = (const float*)d_in[12];
  const float* beta_s = (const float*)d_in[13];
  const float* Wq_e   = (const float*)d_in[14];
  const float* bq_e   = (const float*)d_in[15];
  const float* Wk_e   = (const float*)d_in[16];
  const float* bk_e   = (const float*)d_in[17];
  const float* Wv_e   = (const float*)d_in[18];
  const float* bv_e   = (const float*)d_in[19];
  const float* Wo_e   = (const float*)d_in[20];
  const float* bo_e   = (const float*)d_in[21];
  const float* g_e    = (const float*)d_in[22];
  const float* beta_e = (const float*)d_in[23];
  const float* Wa     = (const float*)d_in[24];
  const float* ba     = (const float*)d_in[25];
  const float* W1     = (const float*)d_in[26];
  const float* b1     = (const float*)d_in[27];
  const float* W2     = (const float*)d_in[28];
  const float* b2     = (const float*)d_in[29];
  const float* g_f    = (const float*)d_in[30];
  const float* beta_f = (const float*)d_in[31];
  const float* Wout   = (const float*)d_in[32];

  // ---- workspace bump allocator ----
  char* base = (char*)d_ws;
  size_t off = 0;
  auto alloc = [&](size_t bytes) -> void* {
    off = (off + 255) & ~(size_t)255;
    void* p = base + off;
    off += bytes;
    return p;
  };
  auto allocF = [&](size_t n) { return (float*)alloc(n * sizeof(float)); };
  auto allocH = [&](size_t n) { return (__bf16*)alloc(n * sizeof(__bf16)); };

  float* tokpos  = allocF((size_t)(cS + 1) * cD);
  float* featpos = allocF((size_t)cNF * cD);
  float* feats   = allocF((size_t)cFR * cD);
  float* xbuf    = allocF((size_t)cBT * cD);
  float* qbuf    = allocF((size_t)cBT * cD);
  float* kbuf    = allocF((size_t)cBT * cD);
  float* vbuf    = allocF((size_t)cBT * cD);
  float* proj    = allocF((size_t)cBT * cD);
  float* sabuf   = allocF((size_t)cBT * cD);
  float* qebuf   = allocF((size_t)cBT * cD);
  float* attnout = allocF((size_t)cBT * cD);
  float* attnbuf = allocF((size_t)cBT * cD);
  float* ff2buf  = allocF((size_t)cBT * cD);
  float* outbuf  = allocF((size_t)cBT * cD);
  float* kebuf   = allocF((size_t)cFR * cD);
  float* vebuf   = allocF((size_t)cFR * cD);
  float* sc_s    = allocF((size_t)cB * cH * cS * cS);
  float* sc_e    = allocF((size_t)cB * cNE * cH * cS * cNF);
  float* av_e    = allocF((size_t)cER * cD);
  float* oebuf   = allocF((size_t)cER * cD);
  float* encbuf  = allocF((size_t)cER * cD);
  float* alphab  = allocF((size_t)cNE * cBT * cD);
  float* catbuf  = allocF((size_t)cNE * cBT * 2 * cD);
  float* ff1buf  = allocF((size_t)cBT * cFF);

  __bf16* wqs  = allocH((size_t)cL * cD * cD);
  __bf16* wks  = allocH((size_t)cL * cD * cD);
  __bf16* wvs  = allocH((size_t)cL * cD * cD);
  __bf16* wos  = allocH((size_t)cL * cD * cD);
  __bf16* wqe  = allocH((size_t)cL * cD * cD);
  __bf16* wke  = allocH((size_t)cL * cD * cD);
  __bf16* wve  = allocH((size_t)cL * cD * cD);
  __bf16* woe  = allocH((size_t)cL * cD * cD);
  __bf16* wab  = allocH((size_t)cL * cNE * 2 * cD * cD);
  __bf16* w1b  = allocH((size_t)cL * cD * cFF);
  __bf16* w2b  = allocH((size_t)cL * cFF * cD);
  __bf16* woub = allocH((size_t)cD * cV);

  auto cvt = [&](const float* W, __bf16* Bt, int mats, int K, int N) {
    const long total = (long)mats * N * K;
    md_convert_wt<<<(int)((total + 255) / 256), 256, 0, stream>>>(W, Bt, K, N, total);
  };
  auto gemm = [&](const float* A, const __bf16* Bt, const float* bias, float* C,
                  int M, int N, int K, int batch,
                  long sA, long sB, long sBias, long sC, int act) {
    dim3 g((N + GBN - 1) / GBN, (M + GBM - 1) / GBM, batch);
    md_gemm_kernel<<<g, 256, 0, stream>>>(A, Bt, bias, C, M, N, K, sA, sB, sBias, sC, act);
  };
  auto ew = [&](long total) { return (int)((total + 255) / 256); };

  // ---- one-time (per-launch) prep ----
  cvt(Wq_s, wqs, cL, cD, cD);
  cvt(Wk_s, wks, cL, cD, cD);
  cvt(Wv_s, wvs, cL, cD, cD);
  cvt(Wo_s, wos, cL, cD, cD);
  cvt(Wq_e, wqe, cL, cD, cD);
  cvt(Wk_e, wke, cL, cD, cD);
  cvt(Wv_e, wve, cL, cD, cD);
  cvt(Wo_e, woe, cL, cD, cD);
  cvt(Wa,   wab, cL * cNE, 2 * cD, cD);
  cvt(W1,   w1b, cL, cD, cFF);
  cvt(W2,   w2b, cL, cFF, cD);
  cvt(Wout, woub, 1, cD, cV);

  md_sinusoid<<<ew((long)(cS + 1) * cD), 256, 0, stream>>>(tokpos, cS + 1, cD, 1);
  md_sinusoid<<<ew((long)cNF * cD), 256, 0, stream>>>(featpos, cNF, cD, 0);
  md_feats<<<ew((long)cFR * cD), 256, 0, stream>>>(features, featpos, feats);
  md_embed<<<ew((long)cBT * cD), 256, 0, stream>>>(tok, emb, outbuf);

  // ---- layers ----
  for (int l = 0; l < cL; l++) {
    const __bf16* wqs_l = wqs + (size_t)l * cD * cD;
    const __bf16* wks_l = wks + (size_t)l * cD * cD;
    const __bf16* wvs_l = wvs + (size_t)l * cD * cD;
    const __bf16* wos_l = wos + (size_t)l * cD * cD;
    const __bf16* wqe_l = wqe + (size_t)l * cD * cD;
    const __bf16* wke_l = wke + (size_t)l * cD * cD;
    const __bf16* wve_l = wve + (size_t)l * cD * cD;
    const __bf16* woe_l = woe + (size_t)l * cD * cD;
    const __bf16* wa_l  = wab + (size_t)l * cNE * 2 * cD * cD;
    const __bf16* w1_l  = w1b + (size_t)l * cD * cFF;
    const __bf16* w2_l  = w2b + (size_t)l * cFF * cD;

    // x = out + tok_pos
    md_addpos<<<ew((long)cBT * cD), 256, 0, stream>>>(outbuf, tokpos, tok, xbuf);

    // self attention
    gemm(xbuf, wqs_l, bq_s + l * cD, qbuf, cBT, cD, cD, 1, 0, 0, 0, 0, 0);
    gemm(xbuf, wks_l, bk_s + l * cD, kbuf, cBT, cD, cD, 1, 0, 0, 0, 0, 0);
    gemm(xbuf, wvs_l, bv_s + l * cD, vbuf, cBT, cD, cD, 1, 0, 0, 0, 0, 0);
    md_self_scores<<<ew((long)cB * cH * cS * cS), 256, 0, stream>>>(qbuf, kbuf, tok, sc_s);
    {
      const int rows = cB * cH * cS;
      md_softmax_rows<<<(rows + 7) / 8, 256, 0, stream>>>(sc_s, rows, cS);
    }
    md_self_av<<<ew((long)cBT * cD), 256, 0, stream>>>(sc_s, vbuf, attnout);
    gemm(attnout, wos_l, bo_s + l * cD, proj, cBT, cD, cD, 1, 0, 0, 0, 0, 0);
    md_ln<<<cBT, 256, 0, stream>>>(xbuf, proj, g_s + l * cD, beta_s + l * cD,
                                   sabuf, cD, 0, 0, tok);

    // cross attention over features
    gemm(sabuf, wqe_l, bq_e + l * cD, qebuf, cBT, cD, cD, 1, 0, 0, 0, 0, 0);
    gemm(feats, wke_l, bk_e + l * cD, kebuf, cFR, cD, cD, 1, 0, 0, 0, 0, 0);
    gemm(feats, wve_l, bv_e + l * cD, vebuf, cFR, cD, cD, 1, 0, 0, 0, 0, 0);
    md_cross_scores<<<ew((long)cB * cNE * cH * cS * cNF), 256, 0, stream>>>(
        qebuf, kebuf, fmask, sc_e);
    {
      const int rows = cB * cNE * cH * cS;
      md_softmax_rows<<<(rows + 7) / 8, 256, 0, stream>>>(sc_e, rows, cNF);
    }
    md_cross_av<<<ew((long)cER * cD), 256, 0, stream>>>(sc_e, vebuf, av_e);
    gemm(av_e, woe_l, bo_e + l * cD, oebuf, cER, cD, cD, 1, 0, 0, 0, 0, 0);
    md_ln<<<cER, 256, 0, stream>>>(sabuf, oebuf, g_e + l * cD, beta_e + l * cD,
                                   encbuf, cD, 1, 1, tok);

    // gated mesh combine
    md_cat<<<ew((long)cNE * cBT * 2 * cD), 256, 0, stream>>>(sabuf, encbuf, catbuf);
    gemm(catbuf, wa_l, ba + (size_t)l * cNE * cD, alphab,
         cBT, cD, 2 * cD, cNE,
         (long)cBT * 2 * cD, (long)cD * 2 * cD, (long)cD, (long)cBT * cD, 2);
    md_combine<<<ew((long)cBT * cD), 256, 0, stream>>>(encbuf, alphab, attnbuf);

    // feed-forward + final LN (with pad zeroing)
    gemm(attnbuf, w1_l, b1 + (size_t)l * cFF, ff1buf, cBT, cFF, cD, 1, 0, 0, 0, 0, 1);
    gemm(ff1buf, w2_l, b2 + (size_t)l * cD, ff2buf, cBT, cD, cFF, 1, 0, 0, 0, 0, 0);
    md_ln<<<cBT, 256, 0, stream>>>(attnbuf, ff2buf, g_f + l * cD, beta_f + l * cD,
                                   outbuf, cD, 0, 1, tok);
  }

  // ---- output projection + log_softmax ----
  float* logits = (float*)d_out;
  gemm(outbuf, woub, nullptr, logits, cBT, cV, cD, 1, 0, 0, 0, 0, 0);
  md_logsoftmax<<<cBT, 256, 0, stream>>>(logits, cV);
}
